// _RPN_distil_residual_20873541059189
// MI455X (gfx1250) — matrix operand order repaired and checked by reading, not compiled, not run on hardware
//
#include <hip/hip_runtime.h>
#include <cstdint>

typedef __attribute__((ext_vector_type(16))) __bf16 v16bf;
typedef __attribute__((ext_vector_type(8)))  __bf16 v8bf;
typedef __attribute__((ext_vector_type(8)))  float  v8f;
typedef __attribute__((ext_vector_type(4)))  int    v4i;

#define AS_GLOBAL __attribute__((address_space(1)))
#define AS_LDS    __attribute__((address_space(3)))

#define NB    8
#define CIN   1024
#define FH    38
#define FW    63
#define OC1   512
#define PADH  40
#define PADW  65
#define NPOS  (FH*FW)          /* 2394  */
#define MTOT  (NB*NPOS)        /* 19152 */
#define KTOT  (CIN*9)          /* 9216  */
#define KB1   (KTOT/32)        /* 288   */
#define MT    (MTOT/16)        /* 1197  */
#define NT1   (OC1/16)         /* 32    */
#define NC2   64
#define NT2   (NC2/16)         /* 4     */
#define KB2   (OC1/32)         /* 16    */
#define NANCH 9
#define NPROP (NPOS*NANCH)     /* 21546 */
#define NSORT 32768
#define PRE_N 2000
#define POST_N 300

#define MG1   9                 /* M-tiles per block in GEMM1 (1197 = 133*9) */
#define NGRP  (OC1/64)          /* 8 N-groups of 64 columns */
#define BTILE (64*32*2)         /* 4 KB staged B tile */

#if __has_builtin(__builtin_amdgcn_global_load_async_to_lds_b128)
#define HAVE_ASYNC_LDS 1
#else
#define HAVE_ASYNC_LDS 0
#endif

__device__ __forceinline__ void wait_asynccnt0() {
#if __has_builtin(__builtin_amdgcn_s_wait_asynccnt)
  __builtin_amdgcn_s_wait_asynccnt(0);
#else
  asm volatile("s_wait_asynccnt 0x0" ::: "memory");
#endif
}

__device__ __forceinline__ void stage16(const __bf16* gsrc, char* ldst) {
#if HAVE_ASYNC_LDS
  __builtin_amdgcn_global_load_async_to_lds_b128(
      (AS_GLOBAL v4i*)gsrc, (AS_LDS v4i*)ldst, 0, 0);
#else
  *(v8bf*)ldst = *(const v8bf*)gsrc;
#endif
}

// Base anchors from generate_anchors(16,(0.5,1,2),(8,16,32)), precomputed exactly.
__device__ __constant__ float c_anch[9][4] = {
  {-84.f,-40.f,99.f,55.f},   {-176.f,-88.f,191.f,103.f}, {-360.f,-184.f,375.f,199.f},
  {-56.f,-56.f,71.f,71.f},   {-120.f,-120.f,135.f,135.f},{-248.f,-248.f,263.f,263.f},
  {-36.f,-80.f,51.f,95.f},   {-80.f,-168.f,95.f,183.f},  {-168.f,-344.f,183.f,359.f}};

// NCHW f32 -> zero-padded NHWC bf16 [8][40][65][1024]
__global__ void k_pad(const float* __restrict__ src, __bf16* __restrict__ dst) {
  size_t t = (size_t)blockIdx.x * blockDim.x + threadIdx.x;
  int c  = (int)(t & (CIN - 1));
  int xy = (int)(t >> 10);
  int x  = xy % PADW;
  int yb = xy / PADW;
  int y  = yb % PADH;
  int b  = yb / PADH;
  float v = 0.f;
  if (y >= 1 && y <= FH && x >= 1 && x <= FW)
    v = src[(((size_t)b * CIN + c) * FH + (y - 1)) * FW + (x - 1)];
  dst[t] = (__bf16)v;
}

// W_conv OIHW f32 -> bf16 blocked [k/32][n][32], k = (r*3+s)*1024 + c
__global__ void k_w1(const float* __restrict__ w, __bf16* __restrict__ dst) {
  int t = blockIdx.x * blockDim.x + threadIdx.x;
  int n = t / KTOT;
  int k = t - n * KTOT;
  int rs = k >> 10;
  int c  = k & 1023;
  int r = rs / 3, s = rs - 3 * r;
  float v = w[(((size_t)n * CIN + c) * 3 + r) * 3 + s];
  dst[((size_t)(k >> 5) * OC1 + n) * 32 + (k & 31)] = (__bf16)v;
}

// Combined 1x1 weights (18 cls + 36 bbox + 10 zero) -> [k/32][64][32] bf16, + bias
__global__ void k_w2(const float* __restrict__ wc, const float* __restrict__ bc,
                     const float* __restrict__ wb, const float* __restrict__ bb,
                     __bf16* __restrict__ dst, float* __restrict__ bias2) {
  int t = blockIdx.x * blockDim.x + threadIdx.x;
  if (t < NC2 * OC1) {
    int n = t & 63;
    int k = t >> 6;
    float v = 0.f;
    if (n < 18) v = wc[n * OC1 + k];
    else if (n < 54) v = wb[(n - 18) * OC1 + k];
    dst[((size_t)(k >> 5) * NC2 + n) * 32 + (k & 31)] = (__bf16)v;
  } else if (t < NC2 * OC1 + NC2) {
    int n = t - NC2 * OC1;
    bias2[n] = (n < 18) ? bc[n] : ((n < 54) ? bb[n - 18] : 0.f);
  }
}

// Implicit-im2col GEMM1: (19152 x 9216) x (9216 x 512), bias+ReLU, bf16 out.
// Block = 9 waves = 9 M-tiles x 64 N cols. B tile (4KB) double-buffered in LDS,
// staged with async-to-LDS. Each wave: 4 accumulators (16x64 output).
__global__ void __launch_bounds__(32*MG1)
k_gemm1(const __bf16* __restrict__ act, const __bf16* __restrict__ w,
        const float* __restrict__ bias, __bf16* __restrict__ out) {
  __shared__ __align__(16) char sbuf[2][BTILE];
  const int t    = threadIdx.x;
  const int wv   = t >> 5;          // 0..8
  const int lane = t & 31;
  const int l15  = lane & 15;
  const int ng   = blockIdx.x;      // 0..7  (64-col group)
  const int mg   = blockIdx.y;      // 0..132
  const int mt   = mg * MG1 + wv;   // 0..1196, always valid
  const int n0   = ng * 64;

  const int m  = mt * 16 + l15;
  const int b  = m / NPOS, pos = m - b * NPOS;
  const int oh = pos / FW, ow = pos - oh * FW;
  const int kA = (lane < 16) ? 0 : 16;
  const int kB = (lane < 16) ? 0 : 16;

  // staged B source: for K-step kb, contiguous 4KB = cols [n0,n0+64) x 32 k
  const __bf16* wbase = w + ((size_t)n0 << 5);

  v8f acc0 = {}, acc1 = {}, acc2 = {}, acc3 = {};

  // prologue: stage tile kb=0 into buffer 0
  if (t < 256)
    stage16(wbase + (size_t)t * 8, &sbuf[0][(size_t)t * 16]);
  wait_asynccnt0();
  __syncthreads();

  int cur = 0;
  #pragma clang loop unroll(disable)
  for (int kb = 0; kb < KB1; ++kb) {
    // issue next tile while computing this one
    if (kb + 1 < KB1 && t < 256)
      stage16(wbase + (((size_t)(kb + 1) * OC1) << 5) + (size_t)t * 8,
              &sbuf[cur ^ 1][(size_t)t * 16]);

    // A fragment (this wave's 16 rows, 32 k)
    int rs = kb >> 5;                 // 32 k-steps per (r,s) since 1024/32==32
    int cb = (kb & 31) << 5;
    int r = rs / 3, s = rs - 3 * r;
    const __bf16* pA = act + ((((size_t)b * PADH + oh + r) * PADW + ow + s) << 10) + cb;
    v8bf alo = *(const v8bf*)(pA + kA);
    v8bf ahi = *(const v8bf*)(pA + kA + 8);
    v16bf af = __builtin_shufflevector(alo, ahi, 0,1,2,3,4,5,6,7,8,9,10,11,12,13,14,15);

    // 4 B fragments from LDS (shared by all 9 waves)
    const __bf16* sb = (const __bf16*)(sbuf[cur]);
    v16bf bf0 = *(const v16bf*)(sb + ((0*16 + l15) << 5) + kB);
    v16bf bf1 = *(const v16bf*)(sb + ((1*16 + l15) << 5) + kB);
    v16bf bf2 = *(const v16bf*)(sb + ((2*16 + l15) << 5) + kB);
    v16bf bf3 = *(const v16bf*)(sb + ((3*16 + l15) << 5) + kB);

    acc0 = __builtin_amdgcn_wmma_f32_16x16x32_bf16(false, af, false, bf0, (short)0, acc0, false, false);
    acc1 = __builtin_amdgcn_wmma_f32_16x16x32_bf16(false, af, false, bf1, (short)0, acc1, false, false);
    acc2 = __builtin_amdgcn_wmma_f32_16x16x32_bf16(false, af, false, bf2, (short)0, acc2, false, false);
    acc3 = __builtin_amdgcn_wmma_f32_16x16x32_bf16(false, af, false, bf3, (short)0, acc3, false, false);

    wait_asynccnt0();
    __syncthreads();
    cur ^= 1;
  }

  const int radd = (lane >> 4) << 3;  // C layout: row = vgpr + 8*(lane>=16), col = lane&15
  #pragma unroll
  for (int sN = 0; sN < 4; ++sN) {
    v8f a = (sN == 0) ? acc0 : (sN == 1) ? acc1 : (sN == 2) ? acc2 : acc3;
    int n = n0 + sN * 16 + l15;
    float bv = bias[n];
    #pragma unroll
    for (int r2 = 0; r2 < 8; ++r2) {
      float v = a[r2] + bv;
      v = v > 0.f ? v : 0.f;
      out[((size_t)(mt * 16 + radd + r2) << 9) + n] = (__bf16)v;
    }
  }
}

// 1x1 convs as GEMM: (19152 x 512) x (512 x 64), f32 out with bias
__global__ void k_gemm2(const __bf16* __restrict__ actc, const __bf16* __restrict__ w,
                        const float* __restrict__ bias, float* __restrict__ out) {
  int wid = (int)(((size_t)blockIdx.x * blockDim.x + threadIdx.x) >> 5);
  if (wid >= MT * NT2) return;
  int lane = threadIdx.x & 31;
  int mt = wid >> 2;          // NT2 == 4
  int nt = wid & 3;
  int l15 = lane & 15;
  int m = mt * 16 + l15;
  int kA = (lane < 16) ? 0 : 16;
  int kB = (lane < 16) ? 0 : 16;
  int n = nt * 16 + l15;
  v8f acc = {};
  for (int kb = 0; kb < KB2; ++kb) {
    const __bf16* pA = actc + ((size_t)m << 9) + (kb << 5);
    v8bf alo = *(const v8bf*)(pA + kA);
    v8bf ahi = *(const v8bf*)(pA + kA + 8);
    const __bf16* pB = w + (((size_t)kb * NC2 + n) << 5);
    __builtin_prefetch(pB + ((size_t)NC2 << 5), 0, 1);
    v16bf bf = *(const v16bf*)(pB + kB);
    v16bf af = __builtin_shufflevector(alo, ahi, 0,1,2,3,4,5,6,7,8,9,10,11,12,13,14,15);
    acc = __builtin_amdgcn_wmma_f32_16x16x32_bf16(false, af, false, bf,
                                                  (short)0, acc, false, false);
  }
  float bv = bias[n];
  int radd = (lane >> 4) << 3;
  #pragma unroll
  for (int r2 = 0; r2 < 8; ++r2)
    out[(size_t)(mt * 16 + radd + r2) * NC2 + n] = acc[r2] + bv;
}

// Softmax score + anchor decode + clip + sort-key pack
__global__ void k_prop(const float* __restrict__ out2, const float* __restrict__ iminfo,
                       float* __restrict__ boxes, unsigned long long* __restrict__ keys) {
  int t = blockIdx.x * blockDim.x + threadIdx.x;
  int b = t >> 15;
  int i = t & (NSORT - 1);
  if (i >= NPROP) { keys[t] = ~0ULL; return; }   // pad entries sort last
  int pos = i / NANCH, a = i - pos * NANCH;
  int oh = pos / FW, ow = pos - oh * FW;
  const float* row = out2 + ((size_t)b * NPOS + pos) * NC2;
  float s0 = row[a], s1 = row[9 + a];
  float mx = fmaxf(s0, s1);
  float e0 = expf(s0 - mx), e1 = expf(s1 - mx);
  float score = e1 / (e0 + e1);
  float dx = row[18 + a*4 + 0], dy = row[18 + a*4 + 1];
  float dw = row[18 + a*4 + 2], dh = row[18 + a*4 + 3];
  float ax1 = c_anch[a][0] + ow * 16.f, ay1 = c_anch[a][1] + oh * 16.f;
  float ax2 = c_anch[a][2] + ow * 16.f, ay2 = c_anch[a][3] + oh * 16.f;
  float w = ax2 - ax1 + 1.f, h = ay2 - ay1 + 1.f;
  float cx = ax1 + 0.5f * w, cy = ay1 + 0.5f * h;
  float px = dx * w + cx, py = dy * h + cy;
  float pw = expf(dw) * w, ph = expf(dh) * h;
  float imh = iminfo[b*3 + 0] - 1.f, imw = iminfo[b*3 + 1] - 1.f;
  float x1 = fminf(fmaxf(px - 0.5f*pw, 0.f), imw);
  float y1 = fminf(fmaxf(py - 0.5f*ph, 0.f), imh);
  float x2 = fminf(fmaxf(px + 0.5f*pw, 0.f), imw);
  float y2 = fminf(fmaxf(py + 0.5f*ph, 0.f), imh);
  size_t bi = ((size_t)b * NPROP + i) * 4;
  boxes[bi] = x1; boxes[bi+1] = y1; boxes[bi+2] = x2; boxes[bi+3] = y2;
  unsigned u = __float_as_uint(score);
  unsigned ord = (u & 0x80000000u) ? ~u : (u | 0x80000000u);   // monotone f32->u32
  keys[t] = (((unsigned long long)(~ord)) << 32) | (unsigned)i; // asc sort => desc score, asc idx
}

__global__ void k_bitonic(unsigned long long* __restrict__ keys, int k, int j) {
  int t = blockIdx.x * blockDim.x + threadIdx.x;
  int i = t & (NSORT - 1);
  int b = t >> 15;
  int ixj = i ^ j;
  if (ixj <= i) return;
  unsigned long long* base = keys + ((size_t)b << 15);
  unsigned long long a = base[i], c = base[ixj];
  bool up = (i & k) == 0;
  if ((a > c) == up) { base[i] = c; base[ixj] = a; }
}

// Greedy NMS on sorted top-2000, emit kept-then-suppressed (== top_k over -inf mask)
__global__ void k_nms(const unsigned long long* __restrict__ keys,
                      const float* __restrict__ boxes, float* __restrict__ out) {
  __shared__ float sx1[PRE_N], sy1[PRE_N], sx2[PRE_N], sy2[PRE_N], sar[PRE_N];
  __shared__ unsigned char keep[PRE_N];
  int b = blockIdx.x;
  int tid = threadIdx.x;
  for (int j = tid; j < PRE_N; j += blockDim.x) {
    unsigned idx = (unsigned)(keys[((size_t)b << 15) + j] & 0xFFFFFFFFu);
    size_t bi = ((size_t)b * NPROP + idx) * 4;
    float x1 = boxes[bi], y1 = boxes[bi+1], x2 = boxes[bi+2], y2 = boxes[bi+3];
    sx1[j] = x1; sy1[j] = y1; sx2[j] = x2; sy2[j] = y2;
    sar[j] = (x2 - x1 + 1.f) * (y2 - y1 + 1.f);
    keep[j] = 1;
  }
  __syncthreads();
  for (int i = 0; i < PRE_N - 1; ++i) {
    if (keep[i]) {
      float x1 = sx1[i], y1 = sy1[i], x2 = sx2[i], y2 = sy2[i], ar = sar[i];
      for (int j = i + 1 + tid; j < PRE_N; j += blockDim.x) {
        float iw = fmaxf(fminf(x2, sx2[j]) - fmaxf(x1, sx1[j]) + 1.f, 0.f);
        float ih = fmaxf(fminf(y2, sy2[j]) - fmaxf(y1, sy1[j]) + 1.f, 0.f);
        float inter = iw * ih;
        float iou = inter / (ar + sar[j] - inter);
        if (iou > 0.7f) keep[j] = 0;
      }
    }
    __syncthreads();
  }
  if (tid == 0) {
    int cnt = 0;
    for (int j = 0; j < PRE_N && cnt < POST_N; ++j) if (keep[j]) {
      size_t o = ((size_t)b * POST_N + cnt) * 5;
      out[o] = (float)b; out[o+1] = sx1[j]; out[o+2] = sy1[j];
      out[o+3] = sx2[j]; out[o+4] = sy2[j];
      ++cnt;
    }
    for (int j = 0; j < PRE_N && cnt < POST_N; ++j) if (!keep[j]) {
      size_t o = ((size_t)b * POST_N + cnt) * 5;
      out[o] = (float)b; out[o+1] = sx1[j]; out[o+2] = sy1[j];
      out[o+3] = sx2[j]; out[o+4] = sy2[j];
      ++cnt;
    }
    if (b == 0) { out[NB*POST_N*5] = 0.f; out[NB*POST_N*5 + 1] = 0.f; }
  }
}

extern "C" void kernel_launch(void* const* d_in, const int* in_sizes, int n_in,
                              void* d_out, int out_size, void* d_ws, size_t ws_size,
                              hipStream_t stream) {
  const float* base_feat = (const float*)d_in[0];
  const float* im_info   = (const float*)d_in[1];
  const float* W_conv    = (const float*)d_in[4];
  const float* b_conv    = (const float*)d_in[5];
  const float* W_cls     = (const float*)d_in[6];
  const float* b_cls     = (const float*)d_in[7];
  const float* W_bbox    = (const float*)d_in[8];
  const float* b_bbox    = (const float*)d_in[9];
  float* out = (float*)d_out;

  char* p = (char*)d_ws;
  auto carve = [&](size_t bytes) -> char* {
    char* r = p; p += (bytes + 255) & ~(size_t)255; return r;
  };
  __bf16* padAct = (__bf16*)carve((size_t)NB * PADH * PADW * CIN * 2);  // 42.6 MB
  __bf16* w1     = (__bf16*)carve((size_t)KTOT * OC1 * 2);              //  9.4 MB
  __bf16* convb  = (__bf16*)carve((size_t)MTOT * OC1 * 2);              // 19.6 MB
  __bf16* w2     = (__bf16*)carve((size_t)OC1 * NC2 * 2);               // 64 KB
  float*  bias2  = (float*) carve((size_t)NC2 * 4);
  float*  out2   = (float*) carve((size_t)MTOT * NC2 * 4);              //  4.9 MB
  float*  boxes  = (float*) carve((size_t)NB * NPROP * 4 * 4);          //  2.8 MB
  unsigned long long* keys = (unsigned long long*)carve((size_t)NB * NSORT * 8);

  k_pad  <<<(NB*PADH*PADW*CIN)/256, 256, 0, stream>>>(base_feat, padAct);
  k_w1   <<<(OC1*KTOT)/256,        256, 0, stream>>>(W_conv, w1);
  k_w2   <<<(NC2*OC1 + NC2 + 255)/256, 256, 0, stream>>>(W_cls, b_cls, W_bbox, b_bbox, w2, bias2);
  k_gemm1<<<dim3(NGRP, MT/MG1, 1), 32*MG1, 0, stream>>>(padAct, w1, b_conv, convb);
  k_gemm2<<<(MT*NT2*32 + 255)/256, 256, 0, stream>>>(convb, w2, bias2, out2);
  k_prop <<<(NB*NSORT)/256,        256, 0, stream>>>(out2, im_info, boxes, keys);
  for (int k = 2; k <= NSORT; k <<= 1)
    for (int j = k >> 1; j > 0; j >>= 1)
      k_bitonic<<<(NB*NSORT)/256, 256, 0, stream>>>(keys, k, j);
  k_nms<<<NB, 256, 0, stream>>>(keys, boxes, out);
}
